// VisionLanguageAttentionKernel_59622736003252
// MI455X (gfx1250) — matrix operand order repaired and checked by reading, not compiled, not run on hardware
//
#include <hip/hip_runtime.h>
#include <hip/hip_bf16.h>

// ---------------------------------------------------------------------------
// CDNA5 / gfx1250 vision-language cross-attention block.
// All matmuls on the wave32 WMMA path (v_wmma_f32_16x16x32_bf16), fp32 accum.
// Weights are transposed+converted to bf16 once so every GEMM's B-operand is
// K-contiguous (128-bit global loads + 128-bit LDS stores + 128-bit fragment
// loads). Epilogues fuse bias / residual / dual fp32+bf16 stores; v2l
// attention probabilities are softmaxed in place inside d_out.
// ---------------------------------------------------------------------------

typedef __bf16 bf16_t;
typedef __attribute__((ext_vector_type(16))) __bf16 v16bf;
typedef __attribute__((ext_vector_type(8)))  __bf16 bf16x8;
typedef __attribute__((ext_vector_type(8)))  float  v8f;

// ---------------------------------------------------------------------------
// Generic WMMA GEMM:  C[M,N] = A[M,K] x B[K,N] (+bias) (+residual)
//   A_F32  : A is fp32, converted to bf16 while staging LDS (softmax probs)
//   B_TRANS: B(k,n) = Bw[n*ldb + k]  (transposed weights, token-major K/V)
//   blockIdx.z batches (batch, head): base offsets zb*s?b + zh*s?h
//   output rows remapped: orow = (row/rpb)*obr + row%rpb + oro  (for concat)
// Fragment layouts per CDNA5 ISA 7.12.2/7.12.4:
//   A element e  <-> K = (e<8 ? hi*8+e : 16+hi*8+(e-8))
//   B element e  <-> K = same order as A            (hi = lane>>4)
// ---------------------------------------------------------------------------
template<int BM, int BN, int WR, int WC,
         bool A_F32, bool B_TRANS,
         bool HAS_BIAS, bool HAS_RES, bool HAS_OUTF, bool HAS_OUTB>
__global__ __launch_bounds__(WR*WC*32)
void gemm_wmma_kernel(const void* __restrict__ Aptr,
                      const bf16_t* __restrict__ Bw,
                      const float* __restrict__ bias,
                      const float* __restrict__ resid,
                      float* __restrict__ outF,
                      bf16_t* __restrict__ outB,
                      int M, int N, int K,
                      int lda, int ldb, int ldo,
                      int rpb, int obr, int oro,
                      int Hn,
                      long long sAb, long long sAh,
                      long long sBb, long long sBh,
                      long long sOb, long long sOh)
{
  constexpr int BK  = 32;
  constexpr int BKP = BK + 8;        // 80B row stride: 16B aligned, bank-spread
  constexpr int WMt = BM / WR;       // wave tile rows (64)
  constexpr int WNt = BN / WC;       // wave tile cols (32)
  constexpr int TM  = WMt / 16;
  constexpr int TN  = WNt / 16;
  constexpr int NT  = WR * WC * 32;

  static_assert(((BM*BK/8) % NT) == 0, "A chunk count");
  static_assert(((BN*BK/8) % NT) == 0, "B chunk count");

  __shared__ __attribute__((aligned(16))) bf16_t As[BM][BKP];  // row-major
  __shared__ __attribute__((aligned(16))) bf16_t Bs[BN][BKP];  // column-major

  const int tid  = threadIdx.x;
  const int lane = tid & 31;
  const int wave = tid >> 5;
  const int wr   = wave / WC;
  const int wc   = wave % WC;
  const int lrow = lane & 15;
  const int lhi  = lane >> 4;

  const int z  = blockIdx.z;
  const int zb = z / Hn;
  const int zh = z % Hn;
  const size_t abase = (size_t)zb * (size_t)sAb + (size_t)zh * (size_t)sAh;
  const size_t bbase = (size_t)zb * (size_t)sBb + (size_t)zh * (size_t)sBh;
  const size_t obase = (size_t)zb * (size_t)sOb + (size_t)zh * (size_t)sOh;

  const float*  Af = (const float*)Aptr;
  const bf16_t* Ab = (const bf16_t*)Aptr;

  const int m0 = blockIdx.y * BM;
  const int n0 = blockIdx.x * BN;
  const bool fullA = (m0 + BM <= M);

  v8f acc[TM][TN];
  #pragma unroll
  for (int i = 0; i < TM; ++i)
    #pragma unroll
    for (int j = 0; j < TN; ++j)
      #pragma unroll
      for (int e = 0; e < 8; ++e) acc[i][j][e] = 0.0f;

  bf16x8 zer;
  #pragma unroll
  for (int j = 0; j < 8; ++j) zer[j] = (bf16_t)0.0f;

  for (int k0 = 0; k0 < K; k0 += BK) {
    // ---- stage A tile (BM x BK): one uint4/2xfloat4 per chunk of 8 --------
    constexpr int AIT = (BM*BK/8) / NT;
    #pragma unroll
    for (int i = 0; i < AIT; ++i) {
      const int cid = tid + i*NT;
      const int r   = cid >> 2;            // BK/8 == 4 chunks per row
      const int ck  = (cid & 3) * 8;
      const int gr  = m0 + r;
      if (fullA || gr < M) {
        const size_t gi = abase + (size_t)gr * (size_t)lda + (size_t)(k0 + ck);
        if (A_F32) {
          const float4 f0 = *(const float4*)(Af + gi);
          const float4 f1 = *(const float4*)(Af + gi + 4);
          bf16x8 h;
          h[0]=(bf16_t)f0.x; h[1]=(bf16_t)f0.y; h[2]=(bf16_t)f0.z; h[3]=(bf16_t)f0.w;
          h[4]=(bf16_t)f1.x; h[5]=(bf16_t)f1.y; h[6]=(bf16_t)f1.z; h[7]=(bf16_t)f1.w;
          *(bf16x8*)(&As[r][ck]) = h;
        } else {
          *(uint4*)(&As[r][ck]) = *(const uint4*)(Ab + gi);
        }
      } else {
        *(bf16x8*)(&As[r][ck]) = zer;
      }
    }
    // ---- stage B tile into column-major Bs[n][k] --------------------------
    constexpr int BIT = (BN*BK/8) / NT;
    if (B_TRANS) {
      // global is K-contiguous: 128-bit load + 128-bit LDS store
      #pragma unroll
      for (int i = 0; i < BIT; ++i) {
        const int cid = tid + i*NT;
        const int c   = cid >> 2;
        const int kc  = (cid & 3) * 8;
        const int gn  = n0 + c;
        if (gn < N) {
          const size_t gi = bbase + (size_t)gn * (size_t)ldb + (size_t)(k0 + kc);
          *(uint4*)(&Bs[c][kc]) = *(const uint4*)(Bw + gi);
        } else {
          *(bf16x8*)(&Bs[c][kc]) = zer;
        }
      }
    } else {
      // global is N-contiguous: 128-bit load, scatter 8 b16 into columns
      constexpr int NC = BN / 8;
      #pragma unroll
      for (int i = 0; i < BIT; ++i) {
        const int cid = tid + i*NT;
        const int k   = cid / NC;
        const int nc  = (cid % NC) * 8;
        const int gn  = n0 + nc;
        union { uint4 u; bf16_t h[8]; } d;
        if (gn < N) {
          d.u = *(const uint4*)(Bw + bbase + (size_t)(k0 + k) * (size_t)ldb + (size_t)gn);
        } else {
          d.u = make_uint4(0u, 0u, 0u, 0u);
        }
        #pragma unroll
        for (int j = 0; j < 8; ++j) Bs[nc + j][k] = d.h[j];
      }
    }
    // speculative prefetch of next K tile of A (global_prefetch_b8)
    if (k0 + BK < K) {
      int pr = m0 + (tid % BM); if (pr >= M) pr = M - 1;
      const char* pa = (const char*)Aptr +
          (abase + (size_t)pr * (size_t)lda + (size_t)(k0 + BK)) * (A_F32 ? 4 : 2);
      __builtin_prefetch(pa, 0, 1);
    }
    __syncthreads();

    // ---- fragments + WMMA (contiguous 16B LDS loads) ----------------------
    union V16 { v16bf v; bf16x8 h[2]; };
    V16 afrag[TM];
    #pragma unroll
    for (int tm = 0; tm < TM; ++tm) {
      const int r = wr*WMt + tm*16 + lrow;
      afrag[tm].h[0] = *(const bf16x8*)(&As[r][lhi*8]);        // K = hi*8+0..7
      afrag[tm].h[1] = *(const bf16x8*)(&As[r][16 + lhi*8]);   // K = 16+hi*8+0..7
    }
    #pragma unroll
    for (int tn = 0; tn < TN; ++tn) {
      const int c = wc*WNt + tn*16 + lrow;
      V16 bfrag;
      bfrag.h[0] = *(const bf16x8*)(&Bs[c][lhi*8]);            // K = hi*8+0..7
      bfrag.h[1] = *(const bf16x8*)(&Bs[c][16 + lhi*8]);       // K = 16+hi*8+0..7
      #pragma unroll
      for (int tm = 0; tm < TM; ++tm)
        acc[tm][tn] = __builtin_amdgcn_wmma_f32_16x16x32_bf16(
            false, afrag[tm].v, false, bfrag.v, (short)0, acc[tm][tn], false, false);
    }
    __syncthreads();
  }

  // ---- epilogue: bias + residual + fp32/bf16 stores (row-remapped) --------
  #pragma unroll
  for (int tm = 0; tm < TM; ++tm) {
    const int mbase = m0 + wr*WMt + tm*16;
    // rpb is a multiple of 16 and mbase is 16-aligned -> uniform per subtile
    const int rq = mbase / rpb;
    const int rr = mbase % rpb;
    const size_t rowbase = (size_t)rq * (size_t)obr + (size_t)rr + (size_t)oro;
    #pragma unroll
    for (int tn = 0; tn < TN; ++tn) {
      const int col = n0 + wc*WNt + tn*16 + lrow;
      const bool cok = (col < N);
      float bv = 0.0f;
      if (HAS_BIAS) { if (cok) bv = bias[col]; }
      #pragma unroll
      for (int e = 0; e < 8; ++e) {
        const int row = mbase + e + 8*lhi;     // C layout: lanes 16-31 -> M+8
        if (row < M && cok) {
          float x = acc[tm][tn][e] + bv;
          const size_t oi = obase + (rowbase + (size_t)(e + 8*lhi)) * (size_t)ldo
                          + (size_t)col;
          if (HAS_RES)  x += resid[oi];
          if (HAS_OUTF) outF[oi] = x;
          if (HAS_OUTB) outB[oi] = (bf16_t)x;
        }
      }
    }
  }
}

// ---------------------------------------------------------------------------
// Transpose + fp32->bf16: o[n*K + k] = (bf16) in[k*N + n]
// 32x32 LDS tiles; K, N multiples of 32.
// ---------------------------------------------------------------------------
__global__ __launch_bounds__(256)
void transpose_cvt_kernel(const float* __restrict__ in, bf16_t* __restrict__ o,
                          int K, int N)
{
  __shared__ float tile[32][33];
  const int bn = blockIdx.x * 32;
  const int bk = blockIdx.y * 32;
  const int tx  = threadIdx.x & 31;
  const int ty4 = (threadIdx.x >> 5) * 4;
  #pragma unroll
  for (int i = 0; i < 4; ++i)
    tile[ty4 + i][tx] = in[(size_t)(bk + ty4 + i) * (size_t)N + (size_t)(bn + tx)];
  __syncthreads();
  #pragma unroll
  for (int i = 0; i < 4; ++i)
    o[(size_t)(bn + ty4 + i) * (size_t)K + (size_t)(bk + tx)] =
        (bf16_t)tile[tx][ty4 + i];
}

// ---------------------------------------------------------------------------
// x + pos, LayerNorm. Emits fp32 residual (x+pos) and bf16 normalized.
// One block per token row of 1024.
// ---------------------------------------------------------------------------
__global__ __launch_bounds__(256)
void addpos_ln_kernel(const float* __restrict__ x, const float* __restrict__ pos,
                      const float* __restrict__ g, const float* __restrict__ b,
                      float* __restrict__ resid, bf16_t* __restrict__ norm,
                      int tokens)
{
  __shared__ float red[256];
  const size_t row  = blockIdx.x;
  const size_t prow = row % (size_t)tokens;   // pos broadcast over batch
  const float* xr = x   + row  * 1024;
  const float* pr = pos + prow * 1024;
  const int tid = threadIdx.x;

  float v[4];
  float s = 0.0f;
  #pragma unroll
  for (int j = 0; j < 4; ++j) { const int c = tid + 256*j; v[j] = xr[c] + pr[c]; s += v[j]; }
  red[tid] = s; __syncthreads();
  for (int t = 128; t > 0; t >>= 1) { if (tid < t) red[tid] += red[tid+t]; __syncthreads(); }
  const float mu = red[0] * (1.0f/1024.0f);
  __syncthreads();

  float sq = 0.0f;
  #pragma unroll
  for (int j = 0; j < 4; ++j) { const float d = v[j] - mu; sq += d*d; }
  red[tid] = sq; __syncthreads();
  for (int t = 128; t > 0; t >>= 1) { if (tid < t) red[tid] += red[tid+t]; __syncthreads(); }
  const float rstd = rsqrtf(red[0] * (1.0f/1024.0f) + 1e-5f);

  #pragma unroll
  for (int j = 0; j < 4; ++j) {
    const int c = tid + 256*j;
    resid[row*1024 + c] = v[j];
    norm [row*1024 + c] = (bf16_t)((v[j] - mu) * rstd * g[c] + b[c]);
  }
}

// ---------------------------------------------------------------------------
// In-place row softmax with scale folded in: p = softmax(scale * logits)
// One 256-thread block per row; L <= 768.
// ---------------------------------------------------------------------------
__global__ __launch_bounds__(256)
void softmax_rows_kernel(float* __restrict__ data, int L, float scale)
{
  __shared__ float red[256];
  float* p = data + (size_t)blockIdx.x * (size_t)L;
  const int tid = threadIdx.x;

  float lv[3];
  int cnt = 0;
  float mx = -3.402823e38f;
  for (int i = tid; i < L; i += 256) { const float t = p[i] * scale; lv[cnt++] = t; mx = fmaxf(mx, t); }
  red[tid] = mx; __syncthreads();
  for (int t = 128; t > 0; t >>= 1) { if (tid < t) red[tid] = fmaxf(red[tid], red[tid+t]); __syncthreads(); }
  mx = red[0]; __syncthreads();

  float s = 0.0f;
  for (int j = 0; j < cnt; ++j) { lv[j] = __expf(lv[j] - mx); s += lv[j]; }
  red[tid] = s; __syncthreads();
  for (int t = 128; t > 0; t >>= 1) { if (tid < t) red[tid] += red[tid+t]; __syncthreads(); }
  const float inv = 1.0f / red[0];

  cnt = 0;
  for (int i = tid; i < L; i += 256) p[i] = lv[cnt++] * inv;
}

// ---------------------------------------------------------------------------
// Host-side orchestration
// ---------------------------------------------------------------------------
extern "C" void kernel_launch(void* const* d_in, const int* in_sizes, int n_in,
                              void* d_out, int out_size, void* d_ws, size_t ws_size,
                              hipStream_t stream)
{
  (void)in_sizes; (void)n_in; (void)out_size; (void)ws_size;

  constexpr int Bc = 8, NP = 576, SL = 512, DM = 1024, NH = 16, HD = 64;
  constexpr float SCALE = 0.125f;                 // 1/sqrt(64)
  const size_t vTok = (size_t)Bc * NP;            // 4608
  const size_t lTok = (size_t)Bc * SL;            // 4096

  const float* vision = (const float*)d_in[0];
  const float* langin = (const float*)d_in[1];
  const float* W32[9] = { (const float*)d_in[2],  (const float*)d_in[4],
                          (const float*)d_in[6],  (const float*)d_in[8],
                          (const float*)d_in[10], (const float*)d_in[12],
                          (const float*)d_in[14], (const float*)d_in[16],
                          (const float*)d_in[18] };
  const float* vq_b  = (const float*)d_in[3];
  const float* vk_b  = (const float*)d_in[5];
  const float* vv_b  = (const float*)d_in[7];
  const float* lq_b  = (const float*)d_in[9];
  const float* lk_b  = (const float*)d_in[11];
  const float* lv_b  = (const float*)d_in[13];
  const float* v2l_b = (const float*)d_in[15];
  const float* l2v_b = (const float*)d_in[17];
  const float* out_b = (const float*)d_in[19];
  const float* vn_g  = (const float*)d_in[20];
  const float* vn_b  = (const float*)d_in[21];
  const float* ln_g  = (const float*)d_in[22];
  const float* ln_b  = (const float*)d_in[23];
  const float* vpos  = (const float*)d_in[24];
  const float* tpos  = (const float*)d_in[25];

  float* out = (float*)d_out;
  const size_t VIS  = (size_t)Bc * (NP + SL) * DM;   // fused ends here
  const size_t LANG = VIS  + vTok * DM;
  const size_t ATTN = LANG + lTok * DM;              // v2l_attn region

  // workspace carve-out (256B aligned)
  size_t cur = 0;
  auto take = [&](size_t bytes) -> void* {
    void* p = (char*)d_ws + cur;
    cur += (bytes + 255) & ~(size_t)255;
    return p;
  };
  bf16_t* wbf[9];                      // TRANSPOSED: wbf[i][n*DM + k]
  for (int i = 0; i < 9; ++i) wbf[i] = (bf16_t*)take((size_t)DM*DM*2);
  float*  visR  = (float*) take(vTok*DM*4);
  float*  langR = (float*) take(lTok*DM*4);
  bf16_t* visN  = (bf16_t*)take(vTok*DM*2);
  bf16_t* langN = (bf16_t*)take(lTok*DM*2);
  bf16_t* vqB   = (bf16_t*)take(vTok*DM*2);
  bf16_t* vkB   = (bf16_t*)take(vTok*DM*2);
  bf16_t* vvB   = (bf16_t*)take(vTok*DM*2);
  bf16_t* lqB   = (bf16_t*)take(lTok*DM*2);
  bf16_t* lkB   = (bf16_t*)take(lTok*DM*2);
  bf16_t* lvB   = (bf16_t*)take(lTok*DM*2);
  float*  l2vLog= (float*) take((size_t)Bc*NH*SL*NP*4);
  bf16_t* vctx  = (bf16_t*)take(vTok*DM*2);
  bf16_t* lctx  = (bf16_t*)take(lTok*DM*2);
  bf16_t* visOB = (bf16_t*)take(vTok*DM*2);
  bf16_t* langOB= (bf16_t*)take(lTok*DM*2);

  // 1) weights fp32 -> bf16, transposed to [N][K] so GEMM B is K-contiguous
  for (int i = 0; i < 9; ++i)
    transpose_cvt_kernel<<<dim3(DM/32, DM/32, 1), 256, 0, stream>>>(W32[i], wbf[i], DM, DM);

  // 2) pos-add + LayerNorm (fp32 residual + bf16 normalized)
  addpos_ln_kernel<<<(int)vTok, 256, 0, stream>>>(vision, vpos, vn_g, vn_b, visR, visN, NP);
  addpos_ln_kernel<<<(int)lTok, 256, 0, stream>>>(langin, tpos, ln_g, ln_b, langR, langN, SL);

  // 3) QKV projections (bf16 out), bias fused.  B_TRANS (weights transposed).
  #define QKV_GEMM gemm_wmma_kernel<128,128,2,4,false,true, true,false,false,true>
  QKV_GEMM<<<dim3(8,36,1),256,0,stream>>>(visN, wbf[0], vq_b, nullptr, nullptr, vqB,
      (int)vTok, DM, DM, DM, DM, DM, (int)vTok, (int)vTok, 0, 1, 0,0, 0,0, 0,0);
  QKV_GEMM<<<dim3(8,36,1),256,0,stream>>>(visN, wbf[1], vk_b, nullptr, nullptr, vkB,
      (int)vTok, DM, DM, DM, DM, DM, (int)vTok, (int)vTok, 0, 1, 0,0, 0,0, 0,0);
  QKV_GEMM<<<dim3(8,36,1),256,0,stream>>>(visN, wbf[2], vv_b, nullptr, nullptr, vvB,
      (int)vTok, DM, DM, DM, DM, DM, (int)vTok, (int)vTok, 0, 1, 0,0, 0,0, 0,0);
  QKV_GEMM<<<dim3(8,32,1),256,0,stream>>>(langN, wbf[3], lq_b, nullptr, nullptr, lqB,
      (int)lTok, DM, DM, DM, DM, DM, (int)lTok, (int)lTok, 0, 1, 0,0, 0,0, 0,0);
  QKV_GEMM<<<dim3(8,32,1),256,0,stream>>>(langN, wbf[4], lk_b, nullptr, nullptr, lkB,
      (int)lTok, DM, DM, DM, DM, DM, (int)lTok, (int)lTok, 0, 1, 0,0, 0,0, 0,0);
  QKV_GEMM<<<dim3(8,32,1),256,0,stream>>>(langN, wbf[5], lv_b, nullptr, nullptr, lvB,
      (int)lTok, DM, DM, DM, DM, DM, (int)lTok, (int)lTok, 0, 1, 0,0, 0,0, 0,0);
  #undef QKV_GEMM

  // 4) logits: Q x K^T per (batch, head). v2l logits land directly in d_out.
  #define LOGIT_GEMM gemm_wmma_kernel<128,128,2,4,false,true, false,false,true,false>
  LOGIT_GEMM<<<dim3(4,5,Bc*NH),256,0,stream>>>(vqB, lkB, nullptr, nullptr, out + ATTN, nullptr,
      NP, SL, HD, DM, DM, SL, NP, NP, 0, NH,
      (long long)NP*DM, HD, (long long)SL*DM, HD,
      (long long)NH*NP*SL, (long long)NP*SL);
  LOGIT_GEMM<<<dim3(5,4,Bc*NH),256,0,stream>>>(lqB, vkB, nullptr, nullptr, l2vLog, nullptr,
      SL, NP, HD, DM, DM, NP, SL, SL, 0, NH,
      (long long)SL*DM, HD, (long long)NP*DM, HD,
      (long long)NH*SL*NP, (long long)SL*NP);
  #undef LOGIT_GEMM

  // 5) softmax (scale folded in), in place
  softmax_rows_kernel<<<Bc*NH*NP, 256, 0, stream>>>(out + ATTN, SL, SCALE);
  softmax_rows_kernel<<<Bc*NH*SL, 256, 0, stream>>>(l2vLog,     NP, SCALE);

  // 6) P x V per (batch, head): fp32 probs converted to bf16 in LDS stage
  #define PV_GEMM gemm_wmma_kernel<256,64,4,2,true,false, false,false,false,true>
  PV_GEMM<<<dim3(1,3,Bc*NH),256,0,stream>>>(out + ATTN, lvB, nullptr, nullptr, nullptr, vctx,
      NP, HD, SL, SL, DM, DM, NP, NP, 0, NH,
      (long long)NH*NP*SL, (long long)NP*SL,
      (long long)SL*DM, HD,
      (long long)NP*DM, HD);
  PV_GEMM<<<dim3(1,2,Bc*NH),256,0,stream>>>(l2vLog, vvB, nullptr, nullptr, nullptr, lctx,
      SL, HD, NP, NP, DM, DM, SL, SL, 0, NH,
      (long long)NH*SL*NP, (long long)SL*NP,
      (long long)NP*DM, HD,
      (long long)SL*DM, HD);
  #undef PV_GEMM

  // 7) context projections: bias + residual(vis/lang), fp32 -> d_out, bf16 copy
  #define CTX_GEMM gemm_wmma_kernel<128,128,2,4,false,true, true,true,true,true>
  CTX_GEMM<<<dim3(8,36,1),256,0,stream>>>(vctx, wbf[6], v2l_b, visR, out + VIS, visOB,
      (int)vTok, DM, DM, DM, DM, DM, (int)vTok, (int)vTok, 0, 1, 0,0, 0,0, 0,0);
  CTX_GEMM<<<dim3(8,32,1),256,0,stream>>>(lctx, wbf[7], l2v_b, langR, out + LANG, langOB,
      (int)lTok, DM, DM, DM, DM, DM, (int)lTok, (int)lTok, 0, 1, 0,0, 0,0, 0,0);
  #undef CTX_GEMM

  // 8) fused = concat(vision_output, language_output) @ out_w + out_b
  //    Row remap implements the concat: vision rows -> b*1088 + i,
  //    language rows -> b*1088 + 576 + i.
  #define OUT_GEMM gemm_wmma_kernel<128,128,2,4,false,true, true,false,true,false>
  OUT_GEMM<<<dim3(8,36,1),256,0,stream>>>(visOB, wbf[8], out_b, nullptr, out, nullptr,
      (int)vTok, DM, DM, DM, DM, DM, NP, NP+SL, 0, 1, 0,0, 0,0, 0,0);
  OUT_GEMM<<<dim3(8,32,1),256,0,stream>>>(langOB, wbf[8], out_b, nullptr, out, nullptr,
      (int)lTok, DM, DM, DM, DM, DM, SL, NP+SL, NP, 1, 0,0, 0,0, 0,0);
  #undef OUT_GEMM
}
